// MHA_12103217840562
// MI455X (gfx1250) — matrix operand order repaired and checked by reading, not compiled, not run on hardware
//
#include <hip/hip_runtime.h>
#include <hip/hip_bf16.h>

// MHA forward for B=4, S=2048, E=1024, NH=16, HD=64 on gfx1250 (CDNA5, wave32).
// QKV GEMM: TDM (tensor_load_to_lds) stages the A tile, v_wmma_f32_16x16x32_f16
// does the math with 32x64 wave tiles. Attention: flash-style, scores never hit HBM.

typedef __attribute__((ext_vector_type(16))) _Float16 v16h;
typedef __attribute__((ext_vector_type(8)))  _Float16 v8h;
typedef __attribute__((ext_vector_type(8)))  float    v8f;
typedef __attribute__((ext_vector_type(4)))  unsigned int su4;
typedef __attribute__((ext_vector_type(8)))  unsigned int su8;

#define NB   4
#define SEQ  2048
#define EMB  1024
#define NHEAD 16
#define HDIM 64
#define MTOT (NB*SEQ)          // 8192 rows
#define NTOT (3*EMB)           // 3072 fused QKV cols
#define PER_BUF ((size_t)NB*NHEAD*SEQ*HDIM)   // 8,388,608 halves per Q/K/V

static __device__ __forceinline__ v16h make16(v8h lo, v8h hi) {
    v16h r;
#pragma unroll
    for (int i = 0; i < 8; ++i) { r[i] = lo[i]; r[i + 8] = hi[i]; }
    return r;
}

// ---------------- Stage 1a: X fp32 -> f16 ----------------
__global__ __launch_bounds__(256) void mha_cvt_x(const float* __restrict__ X,
                                                 _Float16* __restrict__ Xh) {
    size_t i = (size_t)blockIdx.x * 256 + threadIdx.x;
    if (i < (size_t)MTOT * EMB) Xh[i] = (_Float16)X[i];
}

// ---------------- Stage 1b: pack Wq|Wk|Wv -> f16 (EMB x 3072) ----------------
__global__ __launch_bounds__(256) void mha_pack_w(const float* __restrict__ Wq,
                                                  const float* __restrict__ Wk,
                                                  const float* __restrict__ Wv,
                                                  _Float16* __restrict__ Wh) {
    size_t i = (size_t)blockIdx.x * 256 + threadIdx.x;
    if (i >= (size_t)EMB * NTOT) return;
    int e = (int)(i / NTOT);
    int c = (int)(i % NTOT);
    int which = c >> 10;          // 0=Q,1=K,2=V
    int col   = c & 1023;
    float v = (which == 0) ? Wq[(size_t)e * EMB + col]
            : (which == 1) ? Wk[(size_t)e * EMB + col]
                           : Wv[(size_t)e * EMB + col];
    Wh[i] = (_Float16)v;
}

// ---------------- Stage 2: fused QKV GEMM (8192 x 1024 x 3072) ----------------
// block = 128 threads (4 waves); block tile 128x64; wave tile 32x64.
// A tile (128x32 f16) is DMA'd into LDS by the Tensor Data Mover; B tile is
// cooperatively transposed into LDS. 8 WMMAs per wave per 32-wide k-step.
// Outputs: Qh,Kh as (b,h,s,d) f16 ; V transposed as (b,h,d,s) f16.
__global__ __launch_bounds__(128) void mha_qkv_gemm(const _Float16* __restrict__ Xh,
                                                    const _Float16* __restrict__ Wh,
                                                    const float* __restrict__ bq,
                                                    const float* __restrict__ bk,
                                                    const float* __restrict__ bv,
                                                    _Float16* __restrict__ Qh,
                                                    _Float16* __restrict__ Kh,
                                                    _Float16* __restrict__ Vt) {
    __shared__ _Float16 As[128][32];  // A tile, row-major (m,k) — filled by TDM
    __shared__ _Float16 Bt[64][32];   // B tile transposed: Bt[n][k]

    const int tid  = threadIdx.x;
    const int wave = tid >> 5;
    const int lane = tid & 31;
    const int m0 = blockIdx.x * 128;
    const int n0 = blockIdx.y * 64;

    v8f acc[2][4] = {};

    // --- constant parts of the Tensor DMA Descriptor (CDNA5 ISA) ---
    const unsigned int lds_base = (unsigned int)(unsigned long long)(&As[0][0]);
    su8 g1;
    g1[0] = 0x00010000u;                  // workgroup_mask=0, data_size=1 (2 bytes)
    g1[1] = ((unsigned)EMB) << 16;        // tensor_dim0[15:0] @ bits 63:48
    g1[2] = (((unsigned)MTOT & 0xFFFFu) << 16); // dim0 hi16=0 | tensor_dim1 lo16
    g1[3] = (32u << 16);                  // tensor_dim1 hi16=0 | tile_dim0=32
    g1[4] = 128u;                         // tile_dim1=128, tile_dim2=0
    g1[5] = (unsigned)EMB;                // tensor_dim0_stride = 1024 (low 32)
    g1[6] = 0u;                           // stride0 hi16 | dim1_stride lo16
    g1[7] = 0u;

    for (int k0 = 0; k0 < EMB; k0 += 32) {
        // ---- TDM: DMA the 128x32 f16 A tile straight into LDS (wave 0) ----
        if (wave == 0) {
            unsigned long long gaddr =
                (unsigned long long)(Xh + (size_t)m0 * EMB + k0);
            su4 g0;
            g0[0] = 1u;                                   // count=1, user D#
            g0[1] = lds_base;                             // lds_addr
            g0[2] = (unsigned int)gaddr;                  // global_addr[31:0]
            g0[3] = ((unsigned int)(gaddr >> 32) & 0x01FFFFFFu)
                  | 0x80000000u;                          // addr[56:32] | type=2
            asm volatile("tensor_load_to_lds %0, %1"
                         :: "s"(g0), "s"(g1) : "memory");
        }
        // ---- cooperative load of B tile, transposed (coalesced global reads) ----
#pragma unroll
        for (int p = 0; p < 2; ++p) {
            int idx = tid + p * 128;
            int kk = idx >> 3;                  // 0..31
            int c8 = (idx & 7) * 8;             // 0..56
            v8h w = *(const v8h*)(Wh + (size_t)(k0 + kk) * NTOT + n0 + c8);
#pragma unroll
            for (int i = 0; i < 8; ++i) Bt[c8 + i][kk] = w[i];
        }
        if (wave == 0) __builtin_amdgcn_s_wait_tensorcnt(0);
        __syncthreads();

        // ---- fragments & 8 WMMAs (wave tile 32x64) ----
        const int ko = (lane >> 4) * 8;
        v16h a[2];
#pragma unroll
        for (int hb = 0; hb < 2; ++hb) {
            int row = wave * 32 + hb * 16 + (lane & 15);
            v8h alo = *(const v8h*)&As[row][ko];
            v8h ahi = *(const v8h*)&As[row][ko + 16];
            a[hb] = make16(alo, ahi);
        }
#pragma unroll
        for (int j = 0; j < 4; ++j) {
            int col = j * 16 + (lane & 15);
            int kb  = (lane >> 4) * 8;          // lanes<16: K 0..7,16..23 ; lanes>=16: K 8..15,24..31
            v8h blo = *(const v8h*)&Bt[col][kb];
            v8h bhi = *(const v8h*)&Bt[col][kb + 16];
            v16h b = make16(blo, bhi);
            acc[0][j] = __builtin_amdgcn_wmma_f32_16x16x32_f16(
                false, a[0], false, b, (short)0, acc[0][j], false, false);
            acc[1][j] = __builtin_amdgcn_wmma_f32_16x16x32_f16(
                false, a[1], false, b, (short)0, acc[1][j], false, false);
        }
        __syncthreads();
    }

    // epilogue: bias add, convert to f16, scatter into Q/K/V layouts
#pragma unroll
    for (int j = 0; j < 4; ++j) {
        int nl    = n0 + j * 16 + (lane & 15);  // fused column
        int which = nl >> 10;
        int hid   = nl & 1023;
        float bias = (which == 0) ? bq[hid] : (which == 1) ? bk[hid] : bv[hid];
        int h = hid >> 6, d = hid & 63;
#pragma unroll
        for (int hb = 0; hb < 2; ++hb) {
#pragma unroll
            for (int r = 0; r < 8; ++r) {
                int m = m0 + wave * 32 + hb * 16 + r + ((lane >> 4) << 3);
                int b = m >> 11, s = m & 2047;
                _Float16 hv = (_Float16)(acc[hb][j][r] + bias);
                if (which == 0) {
                    Qh[((size_t)(b * NHEAD + h) * SEQ + s) * HDIM + d] = hv;
                } else if (which == 1) {
                    Kh[((size_t)(b * NHEAD + h) * SEQ + s) * HDIM + d] = hv;
                } else {  // V stored transposed: (b,h,d,s)
                    Vt[((size_t)(b * NHEAD + h) * HDIM + d) * SEQ + s] = hv;
                }
            }
        }
    }
}

// ---------------- Stage 3: flash attention ----------------
// grid (S/64, B*NH), block 128 (4 waves); wave owns a 16-row Q tile.
__global__ __launch_bounds__(128) void mha_attn(const _Float16* __restrict__ Qh,
                                                const _Float16* __restrict__ Kh,
                                                const _Float16* __restrict__ Vt,
                                                const int* __restrict__ mask,
                                                float* __restrict__ out) {
    __shared__ _Float16 Ps[4][16][32];   // per-wave P tile (C-layout -> A-layout shuffle)

    const int tid  = threadIdx.x;
    const int wave = tid >> 5;
    const int lane = tid & 31;
    const int bh = blockIdx.y;           // 0..63
    const int b  = bh >> 4;
    const int h  = bh & 15;
    const int q0 = blockIdx.x * 64 + wave * 16;

    const _Float16* Qp = Qh + (size_t)bh * SEQ * HDIM;
    const _Float16* Kp = Kh + (size_t)bh * SEQ * HDIM;
    const _Float16* Vp = Vt + (size_t)bh * HDIM * SEQ;

    // Q A-fragments (contraction dim = 64, split into 2 chunks of 32), kept in regs.
    const int qrow = q0 + (lane & 15);
    const int ko   = (lane >> 4) * 8;
    v16h qa[2];
#pragma unroll
    for (int f = 0; f < 2; ++f) {
        v8h lo = *(const v8h*)(Qp + (size_t)qrow * HDIM + f * 32 + ko);
        v8h hi = *(const v8h*)(Qp + (size_t)qrow * HDIM + f * 32 + ko + 16);
        qa[f] = make16(lo, hi);
    }

    float mrow[8], lrow[8];
#pragma unroll
    for (int r = 0; r < 8; ++r) { mrow[r] = -1e30f; lrow[r] = 0.0f; }
    v8f acc[4] = {};
    const float scale = 0.125f;          // 1/sqrt(64)

    for (int kt = 0; kt < SEQ; kt += 32) {
        if (kt + 32 < SEQ)
            __builtin_prefetch(Kp + (size_t)(kt + 32 + (lane & 31)) * HDIM, 0, 0);

        // ---- scores S = Q K^T  (16 x 32), two 16-col chunks, contraction 64 ----
        v8f sc[2] = {};
#pragma unroll
        for (int j = 0; j < 2; ++j) {
#pragma unroll
            for (int f = 0; f < 2; ++f) {
                int key = kt + j * 16 + (lane & 15);
                int db  = f * 32 + ((lane >> 4) << 3);   // lanes<16 dims 0..7,16..23, >=16 dims 8..15,24..31
                v8h lo = *(const v8h*)(Kp + (size_t)key * HDIM + db);
                v8h hi = *(const v8h*)(Kp + (size_t)key * HDIM + db + 16);
                v16h kb = make16(lo, hi);
                sc[j] = __builtin_amdgcn_wmma_f32_16x16x32_f16(
                    false, qa[f], false, kb, (short)0, sc[j], false, false);
            }
        }

        // ---- scale + mask ----
#pragma unroll
        for (int j = 0; j < 2; ++j) {
            int kc = kt + j * 16 + (lane & 15);
#pragma unroll
            for (int r = 0; r < 8; ++r) {
                int q = q0 + r + ((lane >> 4) << 3);
                int mv = mask[((size_t)b * SEQ + q) * SEQ + kc];
                float v = sc[j][r] * scale;
                sc[j][r] = (mv == 0) ? -1e30f : v;
            }
        }

        // ---- online softmax (per row; row lives on one 16-lane half) ----
#pragma unroll
        for (int r = 0; r < 8; ++r) {
            float mx = fmaxf(sc[0][r], sc[1][r]);
#pragma unroll
            for (int off = 1; off < 16; off <<= 1)
                mx = fmaxf(mx, __shfl_xor(mx, off, 32));
            float nm  = fmaxf(mrow[r], mx);
            float cor = __expf(mrow[r] - nm);
            float p0  = __expf(sc[0][r] - nm);
            float p1  = __expf(sc[1][r] - nm);
            float rs  = p0 + p1;
#pragma unroll
            for (int off = 1; off < 16; off <<= 1)
                rs += __shfl_xor(rs, off, 32);
            lrow[r] = lrow[r] * cor + rs;
            mrow[r] = nm;
#pragma unroll
            for (int j = 0; j < 4; ++j) acc[j][r] *= cor;
            int prow = r + ((lane >> 4) << 3);
            Ps[wave][prow][(lane & 15)]      = (_Float16)p0;
            Ps[wave][prow][16 + (lane & 15)] = (_Float16)p1;
        }
        __syncthreads();

        // ---- P (16x32, A-layout from LDS) x V (32x64) ----
        v8h plo = *(const v8h*)&Ps[wave][lane & 15][ko];
        v8h phi = *(const v8h*)&Ps[wave][lane & 15][ko + 16];
        v16h pa = make16(plo, phi);
#pragma unroll
        for (int j = 0; j < 4; ++j) {
            int d  = j * 16 + (lane & 15);
            int kb = kt + ((lane >> 4) << 3);
            v8h lo = *(const v8h*)(Vp + (size_t)d * SEQ + kb);
            v8h hi = *(const v8h*)(Vp + (size_t)d * SEQ + kb + 16);
            v16h vb = make16(lo, hi);
            acc[j] = __builtin_amdgcn_wmma_f32_16x16x32_f16(
                false, pa, false, vb, (short)0, acc[j], false, false);
        }
        __syncthreads();
    }

    // ---- normalize and store: out[b][q][h*64+d] fp32 ----
#pragma unroll
    for (int j = 0; j < 4; ++j) {
        int d = j * 16 + (lane & 15);
#pragma unroll
        for (int r = 0; r < 8; ++r) {
            int q = q0 + r + ((lane >> 4) << 3);
            float inv = 1.0f / lrow[r];
            out[((size_t)b * SEQ + q) * (NHEAD * HDIM) + h * HDIM + d] = acc[j][r] * inv;
        }
    }
}

extern "C" void kernel_launch(void* const* d_in, const int* in_sizes, int n_in,
                              void* d_out, int out_size, void* d_ws, size_t ws_size,
                              hipStream_t stream) {
    (void)in_sizes; (void)n_in; (void)out_size; (void)ws_size;
    const float* X    = (const float*)d_in[0];
    const int*   mask = (const int*)d_in[1];
    const float* Wq   = (const float*)d_in[2];
    const float* bq   = (const float*)d_in[3];
    const float* Wk   = (const float*)d_in[4];
    const float* bk   = (const float*)d_in[5];
    const float* Wv   = (const float*)d_in[6];
    const float* bv   = (const float*)d_in[7];
    float* out = (float*)d_out;

    char* ws = (char*)d_ws;
    size_t off = 0;
    _Float16* Xh = (_Float16*)(ws + off); off += (size_t)MTOT * EMB * 2;   // 16 MB
    _Float16* Wh = (_Float16*)(ws + off); off += (size_t)EMB * NTOT * 2;   //  6 MB
    _Float16* Qh = (_Float16*)(ws + off); off += PER_BUF * 2;              // 16 MB
    _Float16* Kh = (_Float16*)(ws + off); off += PER_BUF * 2;              // 16 MB
    _Float16* Vt = (_Float16*)(ws + off); off += PER_BUF * 2;              // 16 MB

    mha_cvt_x<<<(unsigned)(((size_t)MTOT * EMB + 255) / 256), 256, 0, stream>>>(X, Xh);
    mha_pack_w<<<(unsigned)(((size_t)EMB * NTOT + 255) / 256), 256, 0, stream>>>(Wq, Wk, Wv, Wh);
    mha_qkv_gemm<<<dim3(MTOT / 128, NTOT / 64), 128, 0, stream>>>(Xh, Wh, bq, bk, bv, Qh, Kh, Vt);
    mha_attn<<<dim3(SEQ / 64, NB * NHEAD), 128, 0, stream>>>(Qh, Kh, Vt, mask, out);
}
